// SelfAttention_89507118448807
// MI455X (gfx1250) — matrix operand order repaired and checked by reading, not compiled, not run on hardware
//
#include <hip/hip_runtime.h>
#include <hip/hip_bf16.h>

typedef __attribute__((ext_vector_type(16))) __bf16        v16bf;
typedef __attribute__((ext_vector_type(8)))  float         v8f;
typedef __attribute__((ext_vector_type(8)))  unsigned int  v8u;
typedef __attribute__((ext_vector_type(4)))  unsigned int  v4u;
typedef __attribute__((ext_vector_type(2)))  unsigned int  v2u;
typedef __attribute__((ext_vector_type(4)))  float         v4f;

#define B_  2
#define N_  2048
#define D_  2048
#define H_  16
#define HD_ 128
#define TD_ 6144      // 3*D
#define M_  4096      // B*N

__device__ __forceinline__ unsigned short f2bf(float f) {
  unsigned int u = __builtin_bit_cast(unsigned int, f);
  return (unsigned short)((u + 0x7FFFu + ((u >> 16) & 1u)) >> 16);
}
__device__ __forceinline__ float bf2f(unsigned short h) {
  unsigned int u = ((unsigned int)h) << 16;
  return __builtin_bit_cast(float, u);
}
// Assemble a v16bf WMMA fragment from two 16-byte chunks (lo, hi).
__device__ __forceinline__ v16bf frag2(const unsigned short* lo, const unsigned short* hi) {
  v4u a = *(const v4u*)lo;
  v4u b = *(const v4u*)hi;
  v8u t;
  t[0] = a[0]; t[1] = a[1]; t[2] = a[2]; t[3] = a[3];
  t[4] = b[0]; t[5] = b[1]; t[6] = b[2]; t[7] = b[3];
  return __builtin_bit_cast(v16bf, t);
}

// CDNA5 async global->LDS copy (16B per lane), tracked by ASYNCcnt.
// LDS address is taken from the vdst VGPR; generic shared ptr low 32 bits = LDS offset.
__device__ __forceinline__ void async_b128(const unsigned short* gsrc,
                                           const unsigned short* ldst) {
  unsigned lds = (unsigned)(unsigned long long)(const void*)ldst;
  unsigned long long ga = (unsigned long long)gsrc;
  asm volatile("global_load_async_to_lds_b128 %0, %1, off"
               :: "v"(lds), "v"(ga) : "memory");
}
__device__ __forceinline__ void wait_async0() {
  asm volatile("s_wait_asynccnt 0x0" ::: "memory");
}

// ---------------------------------------------------------------------------
// Kernel 1: qkv = x @ W^T + b   (f32 in, bf16 out), 128x128 tile / block.
// 8 waves/block, each wave owns a 32x64 accumulator = 8 wmma C tiles.
// K staged 64-deep per barrier pair (16 wmma / wave / stage).
// ---------------------------------------------------------------------------
__global__ __launch_bounds__(256) void qkv_gemm_bf16(
    const float* __restrict__ x, const float* __restrict__ Wq,
    const float* __restrict__ bias, unsigned short* __restrict__ qkv)
{
  __shared__ __align__(16) unsigned short As[128][72];  // bf16, 144B row stride
  __shared__ __align__(16) unsigned short Bs[128][72];

  const int t  = threadIdx.x;
  const int n0 = blockIdx.x * 128;
  const int m0 = blockIdx.y * 128;

  const int w     = t >> 5;
  const int lane  = t & 31;
  const int lrow  = lane & 15;
  const int khalf = lane >> 4;
  const int m_w   = (w & 3) * 32;
  const int n_w   = (w >> 2) * 64;

  const int r0 = t >> 2;        // 0..63 (4 threads per row)
  const int c0 = (t & 3) * 16;  // 0,16,32,48

  v8f zero = {0.f,0.f,0.f,0.f,0.f,0.f,0.f,0.f};
  v8f acc[2][4];
  for (int i = 0; i < 2; ++i)
    for (int j = 0; j < 4; ++j) acc[i][j] = zero;

  for (int k0 = 0; k0 < D_; k0 += 64) {
    // Stage 128x64 A and B tiles, f32 -> bf16 on the way in.
    for (int half = 0; half < 2; ++half) {
      int r = r0 + half * 64;
      const float* xa = &x [(size_t)(m0 + r) * D_ + k0 + c0];
      const float* wb = &Wq[(size_t)(n0 + r) * D_ + k0 + c0];
      if (k0 + 64 < D_) {                 // prefetch next K tile into GL2
        __builtin_prefetch(xa + 64, 0, 3);
        __builtin_prefetch(wb + 64, 0, 3);
      }
      for (int q = 0; q < 4; ++q) {
        v4f av = *(const v4f*)(xa + q * 4);
        v2u pk;
        pk[0] = (unsigned)f2bf(av[0]) | ((unsigned)f2bf(av[1]) << 16);
        pk[1] = (unsigned)f2bf(av[2]) | ((unsigned)f2bf(av[3]) << 16);
        *(v2u*)&As[r][c0 + q * 4] = pk;
        v4f bv = *(const v4f*)(wb + q * 4);
        pk[0] = (unsigned)f2bf(bv[0]) | ((unsigned)f2bf(bv[1]) << 16);
        pk[1] = (unsigned)f2bf(bv[2]) | ((unsigned)f2bf(bv[3]) << 16);
        *(v2u*)&Bs[r][c0 + q * 4] = pk;
      }
    }
    __syncthreads();

    for (int ks = 0; ks < 64; ks += 32) {
      v16bf af[2], bfr[4];
      for (int mf = 0; mf < 2; ++mf) {
        const unsigned short* base = &As[m_w + mf * 16 + lrow][ks];
        af[mf] = frag2(base + khalf * 8, base + 16 + khalf * 8);   // A layout
      }
      for (int nf = 0; nf < 4; ++nf) {
        const unsigned short* base = &Bs[n_w + nf * 16 + lrow][ks];
        bfr[nf] = frag2(base + khalf * 8, base + 16 + khalf * 8);  // B layout
      }
      for (int mf = 0; mf < 2; ++mf)
        for (int nf = 0; nf < 4; ++nf)
          acc[mf][nf] = __builtin_amdgcn_wmma_f32_16x16x32_bf16(
              false, af[mf], false, bfr[nf], (short)0, acc[mf][nf], false, false);
    }
    __syncthreads();
  }

  // Epilogue: + bias, f32 -> bf16, store to qkv workspace.
  for (int mf = 0; mf < 2; ++mf) {
    for (int nf = 0; nf < 4; ++nf) {
      int col = n0 + n_w + nf * 16 + lrow;
      float bv = bias[col];
      int rbase = m0 + m_w + mf * 16 + khalf * 8;
      for (int i = 0; i < 8; ++i)
        qkv[(size_t)(rbase + i) * TD_ + col] = f2bf(acc[mf][nf][i] + bv);
    }
  }
}

// ---------------------------------------------------------------------------
// Kernel 2: interleaved RoPE on first 64 dims of each 128-dim head, q and k.
// ---------------------------------------------------------------------------
__global__ __launch_bounds__(256) void rope_bf16(unsigned short* __restrict__ qkv)
{
  int tid  = blockIdx.x * 256 + threadIdx.x;
  int pair = tid & 31;            // 0..31 -> dims (2p, 2p+1)
  int head = (tid >> 5) & 15;
  int qk   = (tid >> 9) & 1;      // 0 = q, 1 = k
  int row  = tid >> 10;           // 0..4095 = b*N + n
  int pos  = row & (N_ - 1);

  size_t off = (size_t)row * TD_ + (size_t)qk * D_ + head * HD_ + pair * 2;
  unsigned int* p = (unsigned int*)(qkv + off);
  unsigned int v = *p;
  float x1 = bf2f((unsigned short)(v & 0xFFFFu));
  float x2 = bf2f((unsigned short)(v >> 16));
  float inv = __expf(-0.2878231366f * (float)pair);   // 10000^(-2p/64)
  float ang = (float)pos * inv;
  float sn, cs;
  __sincosf(ang, &sn, &cs);
  float o1 = x1 * cs - x2 * sn;
  float o2 = x2 * cs + x1 * sn;
  *p = (unsigned)f2bf(o1) | ((unsigned)f2bf(o2) << 16);
}

// ---------------------------------------------------------------------------
// Kernel 3: causal flash attention. Block = 4 waves, 64 query rows / block.
// K tile staged via async global->LDS (ASYNCcnt), V staged transposed.
// ---------------------------------------------------------------------------
__global__ __launch_bounds__(128) void flash_attn_bf16(
    const unsigned short* __restrict__ qkv, float* __restrict__ out)
{
  __shared__ __align__(16) unsigned short Ks[64][136];     // K: [key][d], 272B stride
  __shared__ __align__(16) unsigned short Vt[128][72];     // V^T: [d][key]
  __shared__ __align__(16) unsigned short Pld[4][16][72];  // per-wave P

  const int t  = threadIdx.x;
  const int qt = blockIdx.x;       // query tile (64 rows)
  const int bh = blockIdx.y;
  const int b  = bh >> 4;
  const int h  = bh & 15;

  const int w     = t >> 5;
  const int lane  = t & 31;
  const int lrow  = lane & 15;
  const int khalf = lane >> 4;

  const unsigned short* qbase = qkv + (size_t)b * N_ * TD_ + (size_t)h * HD_;
  const unsigned short* kbase = qbase + D_;
  const unsigned short* vbase = qbase + 2 * D_;

  const int q0 = qt * 64;

  // Q fragments for this wave's 16 rows (A layout, 4 k-steps of 32).
  v16bf aq[4];
  {
    const unsigned short* qr = qbase + (size_t)(q0 + w * 16 + lrow) * TD_;
    for (int d = 0; d < 4; ++d) {
      const unsigned short* p = qr + d * 32 + khalf * 8;
      aq[d] = frag2(p, p + 16);
    }
  }

  v8f zero = {0.f,0.f,0.f,0.f,0.f,0.f,0.f,0.f};
  v8f Oacc[8];
  for (int i = 0; i < 8; ++i) Oacc[i] = zero;
  float m_run[8], l_run[8];
  for (int i = 0; i < 8; ++i) { m_run[i] = -3.0e38f; l_run[i] = 0.f; }

  const float scale = 0.08838834764831845f;  // 1/sqrt(128)

  for (int kt = 0; kt <= qt; ++kt) {
    __syncthreads();   // previous iteration finished reading Ks/Vt

    // ---- async-stage K tile (64 keys x 128 d bf16 = 16KB), 8 x b128/thread ----
    for (int j = 0; j < 8; ++j) {
      int e   = j * 128 + t;
      int key = e >> 4;            // 16 x 16B chunks per key row
      int ch  = e & 15;
      async_b128(kbase + (size_t)(kt * 64 + key) * TD_ + ch * 8,
                 &Ks[key][ch * 8]);
    }

    // ---- stage V tile transposed; 2 keys/thread so stores pack to b32 ----
    {
      int kp   = t & 31;           // key pair index
      int dseg = t >> 5;           // 0..3 -> 32 d values
      int key0 = kp * 2;
      const unsigned short* v0p = vbase + (size_t)(kt * 64 + key0) * TD_ + dseg * 32;
      const unsigned short* v1p = v0p + TD_;
      for (int c = 0; c < 4; ++c) {
        v4u a  = *(const v4u*)(v0p + c * 8);
        v4u bq = *(const v4u*)(v1p + c * 8);
        for (int e2 = 0; e2 < 4; ++e2) {
          unsigned lo0 = a[e2] & 0xFFFFu,  hi0 = a[e2] >> 16;
          unsigned lo1 = bq[e2] & 0xFFFFu, hi1 = bq[e2] >> 16;
          int d = dseg * 32 + c * 8 + e2 * 2;
          *(unsigned*)&Vt[d][key0]     = lo0 | (lo1 << 16);
          *(unsigned*)&Vt[d + 1][key0] = hi0 | (hi1 << 16);
        }
      }
    }

    wait_async0();     // own async copies landed in LDS
    __syncthreads();   // everyone's copies visible

    // ---- S = Q K^T (16x64 per wave), K from LDS ----
    v8f s[4];
    for (int nt = 0; nt < 4; ++nt) {
      s[nt] = zero;
      const unsigned short* krow = &Ks[nt * 16 + lrow][0];
      for (int d = 0; d < 4; ++d) {
        const unsigned short* p = krow + d * 32 + khalf * 8;
        v16bf bk = frag2(p, p + 16);
        s[nt] = __builtin_amdgcn_wmma_f32_16x16x32_bf16(
            false, aq[d], false, bk, (short)0, s[nt], false, false);
      }
    }

    // ---- scale + causal mask (only on the diagonal tile) ----
    if (kt == qt) {
      for (int nt = 0; nt < 4; ++nt)
        for (int i = 0; i < 8; ++i) {
          int key = kt * 64 + nt * 16 + lrow;
          int qr  = q0 + w * 16 + khalf * 8 + i;
          float v = s[nt][i] * scale;
          s[nt][i] = (key > qr) ? -3.0e38f : v;
        }
    } else {
      for (int nt = 0; nt < 4; ++nt)
        for (int i = 0; i < 8; ++i) s[nt][i] = s[nt][i] * scale;
    }

    // ---- online softmax (rows live across 16 lanes of each half-wave) ----
    float mnew[8];
    for (int i = 0; i < 8; ++i) {
      float mv = m_run[i];
      for (int nt = 0; nt < 4; ++nt) mv = fmaxf(mv, s[nt][i]);
      mnew[i] = mv;
    }
    for (int off = 1; off < 16; off <<= 1)
      for (int i = 0; i < 8; ++i)
        mnew[i] = fmaxf(mnew[i], __shfl_xor(mnew[i], off, 32));

    float alpha[8], psum[8];
    for (int i = 0; i < 8; ++i) {
      alpha[i] = __expf(m_run[i] - mnew[i]);
      m_run[i] = mnew[i];
      psum[i]  = 0.f;
    }
    for (int nt = 0; nt < 4; ++nt)
      for (int i = 0; i < 8; ++i) {
        float p = __expf(s[nt][i] - mnew[i]);
        s[nt][i] = p;
        psum[i] += p;
      }
    for (int off = 1; off < 16; off <<= 1)
      for (int i = 0; i < 8; ++i)
        psum[i] += __shfl_xor(psum[i], off, 32);
    for (int i = 0; i < 8; ++i) l_run[i] = l_run[i] * alpha[i] + psum[i];
    for (int nt = 0; nt < 8; ++nt)
      for (int i = 0; i < 8; ++i) Oacc[nt][i] *= alpha[i];

    // ---- P: C layout -> bf16 -> LDS -> A-fragment layout (same wave) ----
    for (int nt = 0; nt < 4; ++nt)
      for (int i = 0; i < 8; ++i)
        Pld[w][khalf * 8 + i][nt * 16 + lrow] = f2bf(s[nt][i]);

    // ---- O += P V (two 32-deep k-steps over the 64 keys) ----
    for (int ks = 0; ks < 2; ++ks) {
      const unsigned short* pr  = &Pld[w][lrow][0];
      const unsigned short* plo = pr + ks * 32 + khalf * 8;
      v16bf ap = frag2(plo, plo + 16);
      for (int nt = 0; nt < 8; ++nt) {
        const unsigned short* vp = &Vt[nt * 16 + lrow][ks * 32 + khalf * 8];
        v16bf bv = frag2(vp, vp + 16);
        Oacc[nt] = __builtin_amdgcn_wmma_f32_16x16x32_bf16(
            false, ap, false, bv, (short)0, Oacc[nt], false, false);
      }
    }
  }

  // ---- normalize and write y (B, N, D) f32 ----
  for (int i = 0; i < 8; ++i) l_run[i] = 1.0f / l_run[i];
  for (int nt = 0; nt < 8; ++nt) {
    int dcol = h * HD_ + nt * 16 + lrow;
    for (int i = 0; i < 8; ++i) {
      int qr = q0 + w * 16 + khalf * 8 + i;
      out[(size_t)(b * N_ + qr) * D_ + dcol] = Oacc[nt][i] * l_run[i];
    }
  }
}

// ---------------------------------------------------------------------------
extern "C" void kernel_launch(void* const* d_in, const int* in_sizes, int n_in,
                              void* d_out, int out_size, void* d_ws, size_t ws_size,
                              hipStream_t stream)
{
  const float* x    = (const float*)d_in[0];
  const float* Wq   = (const float*)d_in[1];
  const float* bias = (const float*)d_in[2];
  unsigned short* qkv = (unsigned short*)d_ws;   // bf16 (B*N, 3D) = 50.3 MB
  float* out = (float*)d_out;

  dim3 g1(TD_ / 128, M_ / 128);                  // 48 x 32
  qkv_gemm_bf16<<<g1, dim3(256), 0, stream>>>(x, Wq, bias, qkv);

  int rope_threads = M_ * 1024;                  // pairs: B*N * 2(qk) * 16h * 32
  rope_bf16<<<rope_threads / 256, dim3(256), 0, stream>>>(qkv);

  dim3 g3(N_ / 64, B_ * H_);                     // 32 x 32
  flash_attn_bf16<<<g3, dim3(128), 0, stream>>>(qkv, out);
}
